// SSRNet_76673756168879
// MI455X (gfx1250) — matrix operand order repaired and checked by reading, not compiled, not run on hardware
//
#include <hip/hip_runtime.h>
#include <hip/hip_bf16.h>
#include <math.h>

// ---------------- constants (match reference) ----------------
#define BATCH   16384
#define FFIELD  64
#define EDIM    16
#define GROUPS  16
#define DMID    256
#define NLAYER  2
#define LEN1    (FFIELD * EDIM)      // 1024
#define LEN2    (GROUPS * DMID)      // 4096
#define EPSLN   1e-3f

typedef __bf16 v16bf __attribute__((ext_vector_type(16)));
typedef float  v8f   __attribute__((ext_vector_type(8)));
typedef int    v4i_a __attribute__((ext_vector_type(4)));

// ---------------- bf16 helpers (RNE) ----------------
static __device__ __forceinline__ unsigned short f2bf(float f) {
    unsigned int u = __float_as_uint(f);
    unsigned int r = u + 0x7FFFu + ((u >> 16) & 1u);
    return (unsigned short)(r >> 16);
}
static __device__ __forceinline__ float bf2f(unsigned short s) {
    return __uint_as_float(((unsigned int)s) << 16);
}
static __device__ __forceinline__ float gelu_exact(float x) {
    return 0.5f * x * (1.0f + erff(x * 0.70710678118654752f));
}

// ---------------- async global->LDS copy (gfx1250 path w/ fallback) ---------
#if __has_builtin(__builtin_amdgcn_global_load_async_to_lds_b128)
#define ASYNC_LDS 1
#else
#define ASYNC_LDS 0
#endif

static __device__ __forceinline__ void copy_b128_to_lds(
        const unsigned short* g, unsigned short* l) {
#if ASYNC_LDS
    __builtin_amdgcn_global_load_async_to_lds_b128(
        (__attribute__((address_space(1))) v4i_a*)g,
        (__attribute__((address_space(3))) v4i_a*)l, 0, 0);
#else
    *reinterpret_cast<uint4*>(l) = *reinterpret_cast<const uint4*>(g);
#endif
}
static __device__ __forceinline__ void wait_async_copies() {
#if ASYNC_LDS
 #if __has_builtin(__builtin_amdgcn_s_wait_asynccnt)
    __builtin_amdgcn_s_wait_asynccnt(0);
 #else
    asm volatile("s_wait_asynccnt 0" ::: "memory");
 #endif
#endif
}

// ---------------- kernel: f32 -> bf16 weight convert + transpose -------------
// in : W [G, L, D, D]  (d-major rows, e cols)
// out: Wt[G, L, D, D] with Wt[g,l,e,d] = W[g,l,d,e]   (N-major for GEMM B tiles)
__global__ void cvt_bf16_tr_kernel(const float* __restrict__ in,
                                   unsigned short* __restrict__ out) {
    int o = blockIdx.x * blockDim.x + threadIdx.x;     // output-linear
    if (o >= GROUPS * NLAYER * DMID * DMID) return;
    int d  = o & 255;
    int e  = (o >> 8) & 255;
    int gl = o >> 16;
    out[o] = f2bf(in[((size_t)gl * DMID + d) * DMID + e]);
}

// ---------------- kernel: embedding gather * value + LN(E=16) ----------------
__global__ void embed_ln_kernel(const int* __restrict__ feat_index,
                                const float* __restrict__ feat_value,
                                const float* __restrict__ emb_table,
                                const float* __restrict__ gamma_e,
                                const float* __restrict__ beta_e,
                                float* __restrict__ emb_flat) {
    int t = blockIdx.x * blockDim.x + threadIdx.x;     // one (b,f) pair
    if (t >= BATCH * FFIELD) return;
    int row = feat_index[t];
    float val = feat_value[t];
    float v[EDIM];
    float s = 0.f, s2 = 0.f;
    #pragma unroll
    for (int e = 0; e < EDIM; ++e) {
        float x = emb_table[(size_t)row * EDIM + e] * val;
        v[e] = x; s += x; s2 += x * x;
    }
    float mu = s * (1.0f / EDIM);
    float var = s2 * (1.0f / EDIM) - mu * mu;
    float r = rsqrtf(var + EPSLN);
    #pragma unroll
    for (int e = 0; e < EDIM; ++e)
        emb_flat[(size_t)t * EDIM + e] = (v[e] - mu) * r * gamma_e[e] + beta_e[e];
}

// ---------------- kernel: gather columns + cast to bf16 ----------------
__global__ void gather_cast_kernel(const float* __restrict__ src, int S,
                                   const int* __restrict__ idx,
                                   unsigned short* __restrict__ out) {
    long long t = (long long)blockIdx.x * blockDim.x + threadIdx.x;
    if (t >= (long long)BATCH * LEN2) return;
    int b = (int)(t >> 12);
    int j = (int)(t & 4095);
    out[t] = f2bf(src[(size_t)b * S + idx[j]]);
}

// ---------------- kernel: per-group GEMM with bias + exact GELU (bf16 WMMA) ---
// A : bf16 [B, G*D] activations
// Wt: bf16 transposed weights, per-layer base; group stride L*D*D; row e, col d
// bias: per-layer base, group stride L*D.   out: bf16 [B, G*D]
// grid: (M/128, N/64, G), block 256 (8 waves). Wave w owns 16 M-rows.
// Double-buffered LDS, async global->LDS staging.
__global__ __launch_bounds__(256)
void gemm_gelu_kernel(const unsigned short* __restrict__ A,
                      const unsigned short* __restrict__ Wt,
                      const float* __restrict__ bias,
                      unsigned short* __restrict__ out) {
    const int g     = blockIdx.z;
    const int mbase = blockIdx.x * 128;
    const int nbase = blockIdx.y * 64;

    const unsigned short* Ag = A  + (size_t)mbase * LEN2 + g * DMID;
    const unsigned short* Wg = Wt + (size_t)g * (NLAYER * DMID * DMID)
                                  + (size_t)nbase * DMID;           // row = col e
    const float* biasg       = bias + (size_t)g * (NLAYER * DMID);

    __shared__ unsigned short Alds[2][128 * 40];   // rows padded to 40 shorts (80B)
    __shared__ unsigned short Blds[2][64 * 40];    // [col e][k], padded

    const int t      = threadIdx.x;
    const int wave   = t >> 5;
    const int lane   = t & 31;
    const int lanelo = lane & 15;
    const int laneHi = (lane >= 16) ? 1 : 0;

    union AF { v16bf v; unsigned int w[8]; };

    v8f c[4];
    #pragma unroll
    for (int s = 0; s < 4; ++s) c[s] = (v8f){0.f,0.f,0.f,0.f,0.f,0.f,0.f,0.f};

    const int rA  = t >> 2;           // 0..63   (A tile row)
    const int chA = (t & 3) * 8;      // 0,8,16,24 shorts (k chunk)
    const int cB  = t >> 2;           // 0..63   (B tile col e)
    const int chB = (t & 3) * 8;      // k chunk

    // stage one K-step (32 k) of A (128x32) and B (64x32, N-major) into buffer
    auto issue_tiles = [&](int buf, int k0) {
        copy_b128_to_lds(Ag + (size_t)rA        * LEN2 + k0 + chA,
                         &Alds[buf][rA * 40 + chA]);
        copy_b128_to_lds(Ag + (size_t)(rA + 64) * LEN2 + k0 + chA,
                         &Alds[buf][(rA + 64) * 40 + chA]);
        copy_b128_to_lds(Wg + (size_t)cB * DMID + k0 + chB,
                         &Blds[buf][cB * 40 + chB]);
    };

    issue_tiles(0, 0);

    for (int kk = 0; kk < 8; ++kk) {
        wait_async_copies();
        __syncthreads();                       // buffer kk ready for all waves
        if (kk < 7) issue_tiles((kk + 1) & 1, (kk + 1) * 32);
        const int cur = kk & 1;

        // A fragment: lane<16 holds K 0..7 & 16..23, lane>=16 holds +8 offset
        AF af;
        const int rowA = (wave * 16 + lanelo) * 40;
        const int koA  = laneHi ? 8 : 0;
        #pragma unroll
        for (int v = 0; v < 4; ++v)
            af.w[v] = *reinterpret_cast<const unsigned int*>(&Alds[cur][rowA + koA + 2 * v]);
        #pragma unroll
        for (int v = 4; v < 8; ++v)
            af.w[v] = *reinterpret_cast<const unsigned int*>(&Alds[cur][rowA + 16 + koA + 2 * (v - 4)]);

        // B fragments: lane<16 holds K 0..7 & 16..23, lane>=16 holds +8 offset, col = lanelo
        const int koB = laneHi ? 8 : 0;
        #pragma unroll
        for (int s = 0; s < 4; ++s) {
            AF bf;
            const int colB = (s * 16 + lanelo) * 40;
            #pragma unroll
            for (int v = 0; v < 8; ++v)
                bf.w[v] = *reinterpret_cast<const unsigned int*>(&Blds[cur][colB + koB + 2 * (v & 3) + (v >> 2) * 16]);
            c[s] = __builtin_amdgcn_wmma_f32_16x16x32_bf16(
                       false, af.v, false, bf.v, (short)0, c[s], false, false);
        }
    }

    // epilogue: C layout -> lane<16: (M=r, N=lanelo); lane>=16: (M=r+8, N=lanelo)
    const int rowBase = mbase + wave * 16 + (laneHi ? 8 : 0);
    #pragma unroll
    for (int s = 0; s < 4; ++s) {
        const int col = nbase + s * 16 + lanelo;
        const float bv = biasg[col];
        #pragma unroll
        for (int r = 0; r < 8; ++r) {
            float x = c[s][r] + bv;
            out[(size_t)(rowBase + r) * LEN2 + g * DMID + col] = f2bf(gelu_exact(x));
        }
    }
}

// ---------------- kernel: residual (re-gather) + per-group LayerNorm ---------
__global__ __launch_bounds__(256)
void res_ln_kernel(const unsigned short* __restrict__ h2,
                   const float* __restrict__ src, int S,
                   const int* __restrict__ idx,
                   const float* __restrict__ gamma,
                   const float* __restrict__ beta,
                   float* __restrict__ out) {
    const int wave = threadIdx.x >> 5;
    const int lane = threadIdx.x & 31;
    const int p = blockIdx.x * 8 + wave;           // (b,g) pair index
    const int b = p >> 4;
    const int g = p & 15;

    float sv[8];
    float s = 0.f, s2 = 0.f;
    #pragma unroll
    for (int i = 0; i < 8; ++i) {
        const int d = lane + 32 * i;
        float x = src[(size_t)b * S + idx[g * DMID + d]];
        float h = bf2f(h2[(size_t)b * LEN2 + g * DMID + d]);
        float v = h + x;
        sv[i] = v; s += v; s2 += v * v;
    }
    #pragma unroll
    for (int off = 16; off >= 1; off >>= 1) {
        s  += __shfl_xor(s,  off, 32);
        s2 += __shfl_xor(s2, off, 32);
    }
    const float mu  = s * (1.0f / DMID);
    const float var = s2 * (1.0f / DMID) - mu * mu;
    const float r   = rsqrtf(var + EPSLN);
    #pragma unroll
    for (int i = 0; i < 8; ++i) {
        const int d = lane + 32 * i;
        out[(size_t)b * LEN2 + g * DMID + d] =
            (sv[i] - mu) * r * gamma[g * DMID + d] + beta[g * DMID + d];
    }
}

// ---------------- kernel: mean over groups + MLP head -----------------------
__global__ __launch_bounds__(64)
void head_kernel(const float* __restrict__ in,     // [B, G*D]
                 const float* __restrict__ Wp1,    // [256,64]
                 const float* __restrict__ bp1,    // [64]
                 const float* __restrict__ Wp2,    // [64,1]
                 const float* __restrict__ bp2,    // [1]
                 float* __restrict__ outp) {       // [B,1]
    __shared__ float ssr[DMID];
    __shared__ float hid[64];
    const int b = blockIdx.x;
    const int tid = threadIdx.x;

    #pragma unroll
    for (int j = 0; j < 4; ++j) {
        const int d = tid * 4 + j;
        float acc = 0.f;
        #pragma unroll
        for (int g = 0; g < GROUPS; ++g)
            acc += in[(size_t)b * LEN2 + g * DMID + d];
        ssr[d] = acc * (1.0f / GROUPS);
    }
    __syncthreads();

    float a = bp1[tid];
    for (int d = 0; d < DMID; ++d)
        a = fmaf(ssr[d], Wp1[d * 64 + tid], a);
    hid[tid] = fmaxf(a, 0.f);
    __syncthreads();

    if (tid == 0) {
        float sacc = bp2[0];
        #pragma unroll
        for (int j = 0; j < 64; ++j) sacc = fmaf(hid[j], Wp2[j], sacc);
        outp[b] = 1.0f / (1.0f + expf(-sacc));
    }
}

// ---------------- launcher ----------------
extern "C" void kernel_launch(void* const* d_in, const int* in_sizes, int n_in,
                              void* d_out, int out_size, void* d_ws, size_t ws_size,
                              hipStream_t stream) {
    const int*   feat_index = (const int*)  d_in[0];
    const float* feat_value = (const float*)d_in[1];
    const float* emb_table  = (const float*)d_in[2];
    const float* gamma_e    = (const float*)d_in[3];
    const float* beta_e     = (const float*)d_in[4];
    const int*   idx1       = (const int*)  d_in[5];
    const float* W1         = (const float*)d_in[6];
    const float* b1         = (const float*)d_in[7];
    const float* g1         = (const float*)d_in[8];
    const float* be1        = (const float*)d_in[9];
    const int*   idx2       = (const int*)  d_in[10];
    const float* W2         = (const float*)d_in[11];
    const float* b2         = (const float*)d_in[12];
    const float* g2         = (const float*)d_in[13];
    const float* be2        = (const float*)d_in[14];
    const float* Wp1        = (const float*)d_in[15];
    const float* bp1        = (const float*)d_in[16];
    const float* Wp2        = (const float*)d_in[17];
    const float* bp2        = (const float*)d_in[18];
    float* outp = (float*)d_out;

    char* ws = (char*)d_ws;
    const size_t NW = (size_t)GROUPS * NLAYER * DMID * DMID;   // 2,097,152
    size_t off = 0;
    float*          emb_flat = (float*)(ws + off);          off += (size_t)BATCH * LEN1 * 4;   // 64 MB
    unsigned short* W1bf     = (unsigned short*)(ws + off); off += NW * 2;                     // 4 MB
    unsigned short* W2bf     = (unsigned short*)(ws + off); off += NW * 2;                     // 4 MB
    unsigned short* xg       = (unsigned short*)(ws + off); off += (size_t)BATCH * LEN2 * 2;   // 128 MB
    unsigned short* h1       = (unsigned short*)(ws + off); off += (size_t)BATCH * LEN2 * 2;
    unsigned short* h2       = (unsigned short*)(ws + off); off += (size_t)BATCH * LEN2 * 2;
    float*          o1       = (float*)(ws + off);          off += (size_t)BATCH * LEN2 * 4;   // 256 MB
    float*          o2       = (float*)(ws + off);          off += (size_t)BATCH * LEN2 * 4;

    // 1) weight conversion to bf16, transposed to N-major for B-tile staging
    cvt_bf16_tr_kernel<<<(int)((NW + 255) / 256), 256, 0, stream>>>(W1, W1bf);
    cvt_bf16_tr_kernel<<<(int)((NW + 255) / 256), 256, 0, stream>>>(W2, W2bf);

    // 2) embedding + input LN
    embed_ln_kernel<<<(BATCH * FFIELD) / 256, 256, 0, stream>>>(
        feat_index, feat_value, emb_table, gamma_e, beta_e, emb_flat);

    dim3 ggrid((BATCH * LEN2) / 256);
    dim3 mgrid(BATCH / 128, DMID / 64, GROUPS);

    // 3) block 1
    gather_cast_kernel<<<ggrid, 256, 0, stream>>>(emb_flat, LEN1, idx1, xg);
    gemm_gelu_kernel<<<mgrid, 256, 0, stream>>>(xg, W1bf,               b1,        h1);
    gemm_gelu_kernel<<<mgrid, 256, 0, stream>>>(h1, W1bf + DMID * DMID, b1 + DMID, h2);
    res_ln_kernel<<<(BATCH * GROUPS) / 8, 256, 0, stream>>>(h2, emb_flat, LEN1, idx1, g1, be1, o1);

    // 4) block 2
    gather_cast_kernel<<<ggrid, 256, 0, stream>>>(o1, LEN2, idx2, xg);
    gemm_gelu_kernel<<<mgrid, 256, 0, stream>>>(xg, W2bf,               b2,        h1);
    gemm_gelu_kernel<<<mgrid, 256, 0, stream>>>(h1, W2bf + DMID * DMID, b2 + DMID, h2);
    res_ln_kernel<<<(BATCH * GROUPS) / 8, 256, 0, stream>>>(h2, o1, LEN2, idx2, g2, be2, o2);

    // 5) head
    head_kernel<<<BATCH, 64, 0, stream>>>(o2, Wp1, bp1, Wp2, bp2, outp);
}
